// AttentionLayer_18777597018199
// MI455X (gfx1250) — compile-verified, matrix operand order checked by reading, not run on hardware
//
#include <hip/hip_runtime.h>
#include <hip/hip_bf16.h>

// AttentionLayer for MI455X (gfx1250, wave32, WMMA).
// B=2048, F=128, E=64, H=4, A=64, H*A=256.
// One 256-thread (8-wave) workgroup per batch element; per-batch working set
// lives in LDS (~217KB of 320KB, bank-conflict-padded); all GEMMs use
// v_wmma_f32_16x16x32_f16 with f32 accumulation; feature tile staged via the
// CDNA5 async-to-LDS DMA path (global_load_async_to_lds_b128 + s_wait_asynccnt).

constexpr int F  = 128;   // fields
constexpr int E  = 64;    // input emb
constexpr int H  = 4;     // heads
constexpr int A  = 64;    // att emb
constexpr int HA = H * A; // 256

// Padded LDS leading dimensions (+16B per row => per-lane bank starts are
// distinct multiples of 4 on the 64-bank LDS => conflict-free b128 gathers).
constexpr int LDX = E + 4;    // f32 X rows   (68 floats  = 272 B)
constexpr int LDW = E + 8;    // f16 64-wide  (72 halves  = 144 B)
constexpr int LDV = F + 8;    // f16 128-wide (136 halves = 272 B)
constexpr int LDS32 = F + 4;  // f32 scores   (132 floats = 528 B)

typedef __attribute__((ext_vector_type(16))) _Float16 v16h;
typedef __attribute__((ext_vector_type(8)))  _Float16 v8h;
typedef __attribute__((ext_vector_type(8)))  float    v8f;
typedef __attribute__((ext_vector_type(4)))  float    v4f;
typedef __attribute__((ext_vector_type(4)))  int      v4i;

// Address-space-qualified element types for the async DMA builtin
// (its params are v4i in AS1/AS3; HIP prints AS1 as `__device__`).
typedef __attribute__((address_space(1))) v4i g_v4i;   // global
typedef __attribute__((address_space(3))) v4i l_v4i;   // LDS

__device__ __forceinline__ v8f wmma16(v16h a, v16h b, v8f c) {
    // 8 args: (neg_a, A, neg_b, B, c_mod, C, reuse_a, reuse_b)
    return __builtin_amdgcn_wmma_f32_16x16x32_f16(false, a, false, b,
                                                  (short)0, c, false, false);
}

// 16x32 f16 fragment (A layout; also B when source is K-contiguous, i.e. B^T).
// ISA 7.12.2: lane L(0-15) row M=L holds K 0-7 / 16-23; lanes 16-31 at K+8.
__device__ __forceinline__ v16h frag_ld16(const _Float16* __restrict__ s,
                                          int ld, int row0, int k0, int lane) {
    const _Float16* p = s + (row0 + (lane & 15)) * ld + k0 + ((lane >> 4) << 3);
    v8h lo = *(const v8h*)(p);
    v8h hi = *(const v8h*)(p + 16);
    return __builtin_shufflevector(lo, hi, 0, 1, 2, 3, 4, 5, 6, 7,
                                   8, 9, 10, 11, 12, 13, 14, 15);
}

// Same fragment shape gathered from an f32 matrix, converting to f16 on the
// fly (v_cvt co-executes with the XDL matrix pipe).
__device__ __forceinline__ v16h frag_ld32(const float* __restrict__ s,
                                          int ld, int row0, int k0, int lane) {
    const float* p = s + (row0 + (lane & 15)) * ld + k0 + ((lane >> 4) << 3);
    v16h a;
#pragma unroll
    for (int i = 0; i < 8; ++i) {
        a[i]     = (_Float16)p[i];
        a[i + 8] = (_Float16)p[16 + i];
    }
    return a;
}

// C/D 16x16 f32 layout: lane L -> col n = L&15, VGPR r -> row m = r + 8*(L>=16).
__device__ __forceinline__ void store_c_f32(float* s, int ld, int m0, int n0,
                                            v8f c, int lane) {
    float* p = s + (m0 + ((lane >> 4) << 3)) * ld + n0 + (lane & 15);
#pragma unroll
    for (int r = 0; r < 8; ++r) p[r * ld] = c[r];
}

__device__ __forceinline__ void store_c_f16(_Float16* s, int ld, int m0, int n0,
                                            v8f c, int lane) {
    _Float16* p = s + (m0 + ((lane >> 4) << 3)) * ld + n0 + (lane & 15);
#pragma unroll
    for (int r = 0; r < 8; ++r) p[r * ld] = (_Float16)c[r];
}

// Transposed f16 store: dst is [N][M] row-major -> one contiguous 16B ds store.
__device__ __forceinline__ void store_c_f16_t(_Float16* s, int ldT, int m0,
                                              int n0, v8f c, int lane) {
    _Float16* p = s + (n0 + (lane & 15)) * ldT + m0 + ((lane >> 4) << 3);
    v8h t;
#pragma unroll
    for (int r = 0; r < 8; ++r) t[r] = (_Float16)c[r];
    *(v8h*)p = t;
}

__global__ __launch_bounds__(256)
void attn_interact_kernel(const float* __restrict__ feature,
                          const float* __restrict__ Qw,
                          const float* __restrict__ Kw,
                          const float* __restrict__ Vw,
                          const float* __restrict__ RW,
                          float* __restrict__ out) {
    // ---- LDS working set (~217 KB of 320 KB WGP LDS) ----
    __shared__ alignas(16) float    sX32[F * LDX];       // 34 KB  X f32 [f][e]
    __shared__ alignas(16) _Float16 sQ  [F * LDW];       // 18 KB  q  [f][a]
    __shared__ alignas(16) _Float16 sKt [F * LDW];       // 18 KB  k  [f][a] (B^T gather)
    __shared__ alignas(16) _Float16 sVt [A * LDV];       // 17 KB  v^T [a][f]
    __shared__ alignas(16) _Float16 sWt [3 * A * LDW];   // 27 KB  Q/K/V[h]^T [a][e]
    __shared__ alignas(16) _Float16 sRWt[HA * LDW];      // 36 KB  res_w^T [c][e]
    __shared__ alignas(16) float    sS  [F * LDS32];     // 66 KB  scores/probs f32
    __shared__ float sMax[F];
    __shared__ float sRcp[F];

    const int tid  = threadIdx.x;
    const int lane = tid & 31;
    // Scalarized wave id: keeps all tile indices in SGPRs -> uniform
    // s_cbranch control flow, EXEC provably all-ones around WMMA.
    const int wv   = __builtin_amdgcn_readfirstlane(tid >> 5);
    const int b    = blockIdx.x;

    const float* xg = feature + (size_t)b * F * E;
    float*       og = out     + (size_t)b * F * HA;

    // ---- Stage X f32 into LDS via async DMA (tracked by ASYNCcnt) ----
    {
        constexpr int CH = (F * E) / 4;        // 2048 16-byte chunks
        for (int c = tid; c < CH; c += 256) {
            int row = c >> 4, col4 = (c & 15) << 2;   // 16 chunks per row
            const float* gp = xg + row * E + col4;
            float* lp = &sX32[row * LDX + col4];
#if defined(__gfx1250__) && __has_builtin(__builtin_amdgcn_global_load_async_to_lds_b128)
            // Generic->AS1 global pointers are bit-identical; generic LDS
            // pointers hold the LDS byte offset in the low 32 bits (ISA
            // aperture rule), so truncation yields the AS3 pointer.
            __builtin_amdgcn_global_load_async_to_lds_b128(
                (g_v4i*)(uintptr_t)gp,
                (l_v4i*)(uint32_t)(uintptr_t)lp,
                /*offset=*/0, /*cpol=*/0);
#else
            *(v4f*)lp = *(const v4f*)gp;
#endif
        }
#if __has_builtin(__builtin_amdgcn_s_wait_asynccnt)
        __builtin_amdgcn_s_wait_asynccnt(0);
#else
        asm volatile("s_wait_asynccnt 0" ::: "memory");
#endif
    }
    // Stage res_w transposed: sRWt[c][e] = RW[e][c] (coalesced read, scattered ds).
    for (int i = tid; i < E * HA; i += 256) {
        int e = i >> 8, c = i & 255;
        sRWt[c * LDW + e] = (_Float16)RW[i];
    }
    __syncthreads();

    for (int h = 0; h < H; ++h) {
        // Stage per-head Q/K/V weights transposed: sWt[m][a][e] = W[h][e][a].
        for (int i = tid; i < E * A; i += 256) {
            int e = i >> 6, a = i & 63;
            int t = a * LDW + e;
            sWt[t]                 = (_Float16)Qw[h * E * A + i];
            sWt[A * LDW + t]       = (_Float16)Kw[h * E * A + i];
            sWt[2 * A * LDW + t]   = (_Float16)Vw[h * E * A + i];
        }
        __syncthreads();

        // q/k/v projections: 3 mats x 8 Mtiles x 4 Ntiles = 96 tiles, 8 waves.
        for (int t = wv; t < 96; t += 8) {
            int mat = t >> 5, rem = t & 31, mt = rem >> 2, nt = rem & 3;
            const _Float16* W = sWt + mat * A * LDW;
            v8f acc = {};
#pragma unroll
            for (int ks = 0; ks < 2; ++ks) {
                v16h af = frag_ld32(sX32, LDX, mt * 16, ks * 32, lane);
                v16h bf = frag_ld16(W,    LDW, nt * 16, ks * 32, lane); // [a][e] => B^T
                acc = wmma16(af, bf, acc);
            }
            if (mat == 0)      store_c_f16 (sQ,  LDW, mt * 16, nt * 16, acc, lane);
            else if (mat == 1) store_c_f16 (sKt, LDW, mt * 16, nt * 16, acc, lane);
            else               store_c_f16_t(sVt, LDV, mt * 16, nt * 16, acc, lane);
        }
        __syncthreads();

        // scores S = q @ k^T : 8x8 tiles over 8 waves, K=64 (2 wmma steps).
        for (int t = wv; t < 64; t += 8) {
            int mt = t >> 3, nt = t & 7;
            v8f acc = {};
#pragma unroll
            for (int ks = 0; ks < 2; ++ks) {
                v16h af = frag_ld16(sQ,  LDW, mt * 16, ks * 32, lane);
                v16h bf = frag_ld16(sKt, LDW, nt * 16, ks * 32, lane);
                acc = wmma16(af, bf, acc);
            }
            store_c_f32(sS, LDS32, mt * 16, nt * 16, acc, lane);
        }
        __syncthreads();

        // Softmax over rows (no 1/sqrt(d) scaling, per reference).
        // Pass 1: paired lanes (tid, tid^1) reduce one row's max & sum.
        {
            int r = tid >> 1, hf = tid & 1;
            const float* row = sS + r * LDS32 + hf * 64;
            float m = -3.4e38f;
            for (int j = 0; j < 64; ++j) m = fmaxf(m, row[j]);
            m = fmaxf(m, __shfl_xor(m, 1));
            float sum = 0.f;
            for (int j = 0; j < 64; ++j) sum += __expf(row[j] - m);
            sum += __shfl_xor(sum, 1);
            if (!hf) { sMax[r] = m; sRcp[r] = 1.0f / sum; }
        }
        __syncthreads();
        // Pass 2: normalize in place (v_exp_f32 is TRANS, co-executes w/ VALU).
        for (int i = tid; i < F * F; i += 256) {
            int r = i >> 7, c = i & 127;
            float* p = sS + r * LDS32 + c;
            *p = __expf(*p - sMax[r]) * sRcp[r];
        }
        __syncthreads();

        // Output tiles: fuse em = S@v (4 wmma) with res = X@res_w (2 wmma),
        // ReLU, and write f32 straight to HBM (single pass over output).
        for (int t = wv; t < 32; t += 8) {
            int mt = t >> 2, nt = t & 3;
            int cg = h * A + nt * 16;     // global output column base
            v8f acc = {};
#pragma unroll
            for (int ks = 0; ks < 2; ++ks) {   // residual branch, K = E = 64
                v16h af = frag_ld32(sX32, LDX, mt * 16, ks * 32, lane);
                v16h bf = frag_ld16(sRWt, LDW, cg,      ks * 32, lane);
                acc = wmma16(af, bf, acc);
            }
#pragma unroll
            for (int ks = 0; ks < 4; ++ks) {   // attention branch, K = F = 128
                v16h af = frag_ld32(sS,  LDS32, mt * 16, ks * 32, lane);
                v16h bf = frag_ld16(sVt, LDV,   nt * 16, ks * 32, lane);
                acc = wmma16(af, bf, acc);
            }
            float* p = og + (mt * 16 + ((lane >> 4) << 3)) * HA + cg + (lane & 15);
#pragma unroll
            for (int r = 0; r < 8; ++r) p[r * HA] = fmaxf(acc[r], 0.f);
        }
        __syncthreads();   // LDS buffers reused next head
    }
}

extern "C" void kernel_launch(void* const* d_in, const int* in_sizes, int n_in,
                              void* d_out, int out_size, void* d_ws, size_t ws_size,
                              hipStream_t stream) {
    const float* feature = (const float*)d_in[0];
    const float* Qw      = (const float*)d_in[1];
    const float* Kw      = (const float*)d_in[2];
    const float* Vw      = (const float*)d_in[3];
    const float* RW      = (const float*)d_in[4];
    float* out           = (float*)d_out;

    const int B = in_sizes[0] / (F * E);   // 2048
    dim3 grid(B), block(256);
    hipLaunchKernelGGL(attn_interact_kernel, grid, block, 0, stream,
                       feature, Qw, Kw, Vw, RW, out);
}
